// SCAB_downsample_20650202759885
// MI455X (gfx1250) — matrix operand order repaired and checked by reading, not compiled, not run on hardware
//
#include <hip/hip_runtime.h>
#include <hip/hip_bf16.h>

// MI455X / gfx1250 fused SCAB-upsample kernel.
// Roofline: ~85 MB mandatory HBM traffic (x in, out out) => ~3.6us @ 23.3 TB/s.
// All dense contractions (conv1x1 layers, offset/route head, expert
// compress/expand einsums) run on v_wmma_f32_16x16x32_f16 (f32 accumulate);
// bilinear gathers stay fp32 and are served out of the 192 MB L2 (x = 16.7 MB).

typedef _Float16 v16h __attribute__((ext_vector_type(16)));
typedef _Float16 v8h  __attribute__((ext_vector_type(8)));
typedef float    v8f  __attribute__((ext_vector_type(8)));

// Intra-wave LDS phase fence: DS ops are in-order per wave; this pins the
// compile-time ordering and drains DScnt before the next phase reads LDS.
#define WAVE_LDS_FENCE() asm volatile("s_wait_dscnt 0" ::: "memory")

__device__ __forceinline__ v16h frag_zip(v8h a0, v8h a1) {
  v16h r;
#pragma unroll
  for (int i = 0; i < 8; ++i) { r[i] = a0[i]; r[i + 8] = a1[i]; }
  return r;
}

// A-matrix fragment (16xK tile, f16): element e of lane l holds
// K = kb + e + 16*(l>=16), M = l&15.
__device__ __forceinline__ v16h loadAfrag(const _Float16* w, int row, int rs,
                                          int kb, int hv) {
  const _Float16* p = w + row * rs + kb + hv * 16;
  return frag_zip(*(const v8h*)p, *(const v8h*)(p + 8));
}

// B-matrix fragment (Kx16 tile, f16): element j of lane l holds
// K = kb + j + 16*(l>=16), N = l&15 (contiguous k order, same as loadAfrag).
__device__ __forceinline__ v16h loadBfrag(const _Float16* act, int n, int rs,
                                          int kb, int hv) {
  const _Float16* p = act + n * rs + kb + hv * 16;
  return frag_zip(*(const v8h*)p, *(const v8h*)(p + 8));
}

__device__ __forceinline__ v8f wmma16(v16h a, v16h b, v8f c) {
  return __builtin_amdgcn_wmma_f32_16x16x32_f16(false, a, false, b, (short)0, c,
                                                false, false);
}

struct Bilerp {
  int o00, o01, o10, o11;
  float w00, w01, w10, w11;
};

__device__ __forceinline__ Bilerp mk_bilerp(float px, float py, int H_, int W_) {
  float xf = floorf(px), yf = floorf(py);
  float fx = px - xf, fy = py - yf;
  int x0 = (int)xf, y0 = (int)yf, x1 = x0 + 1, y1 = y0 + 1;
  bool vx0 = (x0 >= 0) && (x0 < W_), vx1 = (x1 >= 0) && (x1 < W_);
  bool vy0 = (y0 >= 0) && (y0 < H_), vy1 = (y1 >= 0) && (y1 < H_);
  int x0c = min(max(x0, 0), W_ - 1), x1c = min(max(x1, 0), W_ - 1);
  int y0c = min(max(y0, 0), H_ - 1), y1c = min(max(y1, 0), H_ - 1);
  Bilerp t;
  t.o00 = y0c * W_ + x0c; t.o01 = y0c * W_ + x1c;
  t.o10 = y1c * W_ + x0c; t.o11 = y1c * W_ + x1c;
  t.w00 = (vy0 && vx0) ? (1.f - fy) * (1.f - fx) : 0.f;
  t.w01 = (vy0 && vx1) ? (1.f - fy) * fx : 0.f;
  t.w10 = (vy1 && vx0) ? fy * (1.f - fx) : 0.f;
  t.w11 = (vy1 && vx1) ? fy * fx : 0.f;
  return t;
}

__device__ __forceinline__ float sigmoidf_(float v) {
  return 1.f / (1.f + __expf(-v));
}

// B=4, C=64, H=W=128, E=4, scale=2 -> outH=outW=256.
// Each wave owns 16 output pixels (WMMA N). Block = 4 waves = 64 px.
// Grid = 262144 px / 64 = 4096 blocks (exact).
__global__ __launch_bounds__(128, 1) void scab_fused_gfx1250(
    const float* __restrict__ x,     // (4,64,128,128)
    const float* __restrict__ wc,    // (4,8,64)  weight_compress
    const float* __restrict__ we,    // (4,64,8)  weight_expand
    const float* __restrict__ w1,    // (64,68)
    const float* __restrict__ b1,    // (64)
    const float* __restrict__ w2,    // (64,64)
    const float* __restrict__ b2,    // (64)
    const float* __restrict__ wr,    // (4,64)    w_route
    const float* __restrict__ br,    // (4)
    const float* __restrict__ wo,    // (2,64)    w_off
    const float* __restrict__ bo,    // (2)
    float* __restrict__ out)         // (4,64,256,256)
{
  // ---- static LDS (~50 KB total, well under 320 KB/WGP) ----
  __shared__ _Float16 sW1[64 * 96];   // layer1 weights, K padded 68->96
  __shared__ _Float16 sW2[64 * 64];   // layer2 weights
  __shared__ _Float16 sWOR[16 * 64];  // rows 0-1 = w_off, 2-5 = w_route, rest 0
  __shared__ _Float16 sWC[32 * 64];   // expert-stacked compress: row e*8+k
  __shared__ _Float16 sWE[64 * 32];   // expert-stacked expand:   col e*8+k
  __shared__ __align__(16) unsigned char sScratch[4][5120];  // per-wave

  const int tid = threadIdx.x;

  // ---- stage + f32->f16 convert all weights once per block ----
  for (int i = tid; i < 64 * 96; i += 128) {
    int o = i / 96, k = i - o * 96;
    sW1[i] = (_Float16)((k < 68) ? w1[o * 68 + k] : 0.f);
  }
  for (int i = tid; i < 64 * 64; i += 128) sW2[i] = (_Float16)w2[i];
  for (int i = tid; i < 16 * 64; i += 128) {
    int r = i >> 6, k = i & 63;
    float v = (r < 2) ? wo[r * 64 + k] : ((r < 6) ? wr[(r - 2) * 64 + k] : 0.f);
    sWOR[i] = (_Float16)v;
  }
  for (int i = tid; i < 32 * 64; i += 128) sWC[i] = (_Float16)wc[i];
  for (int i = tid; i < 64 * 32; i += 128) {
    int c = i >> 5, ek = i & 31;
    sWE[i] = (_Float16)we[((ek >> 3) * 64 + c) * 8 + (ek & 7)];
  }
  __syncthreads();

  const int wave = tid >> 5, lane = tid & 31;
  const int n = lane & 15;      // pixel column within the wave tile (WMMA N)
  const int hv = lane >> 4;     // lane half: also selects channel half for gathers

  _Float16* bufA = (_Float16*)(sScratch[wave]);          // 16x96 h: input_cat; later 16x64 h: emb2
  _Float16* bufB = (_Float16*)(sScratch[wave] + 3072);   // 16x64 h: emb
  float* feaF = (float*)(sScratch[wave]);                // 16x64 f32: fea0 (aliases bufA/bufB)

  const int p = (blockIdx.x * 4 + wave) * 16 + n;  // flat output pixel
  const int b = p >> 16;
  const int oy = (p >> 8) & 255;
  const int ox = p & 255;

  // ---- base sampling grid (reference formulas, scale=2) ----
  const float inv_s = 0.5f;
  const float gx = ((ox + 0.5f) * inv_s - 0.5f) * (2.0f / 127.f) - 1.0f;
  const float gy = ((oy + 0.5f) * inv_s - 0.5f) * (2.0f / 127.f) - 1.0f;
  const float px = ((gx + 1.0f) * 128.f - 1.0f) * 0.5f;
  const float py = ((gy + 1.0f) * 128.f - 1.0f) * 0.5f;

  const float* xb = x + b * (64 * 128 * 128);

  // ---- pre_fea: lane samples 32 channels of its pixel, writes f16 into LDS
  {
    Bilerp t = mk_bilerp(px, py, 128, 128);
    const float* xc = xb + (hv * 32) * 16384;
#pragma unroll 4
    for (int i = 0; i < 32; ++i) {
      float v = t.w00 * xc[t.o00] + t.w01 * xc[t.o01] +
                t.w10 * xc[t.o10] + t.w11 * xc[t.o11];
      bufA[n * 96 + 4 + hv * 32 + i] = (_Float16)v;
      xc += 16384;
    }
    if (hv == 0) {  // coords occupy k=0..3
      float th = (oy + 0.5f) * inv_s; th = th - floorf(th + 0.001f) - 0.5f;
      float tw = (ox + 0.5f) * inv_s; tw = tw - floorf(tw + 0.001f) - 0.5f;
      bufA[n * 96 + 0] = (_Float16)inv_s;
      bufA[n * 96 + 1] = (_Float16)inv_s;
      bufA[n * 96 + 2] = (_Float16)th;
      bufA[n * 96 + 3] = (_Float16)tw;
    } else {        // zero-fill K pad 68..95
      for (int k = 68; k < 96; ++k) bufA[n * 96 + k] = (_Float16)0.f;
    }
  }
  WAVE_LDS_FENCE();

  // ---- layer1: emb = relu(W1 @ input_cat + b1)  (M=64, K=96, N=16) ----
  {
    v16h bf0 = loadBfrag(bufA, n, 96, 0, hv);
    v16h bf1 = loadBfrag(bufA, n, 96, 32, hv);
    v16h bf2 = loadBfrag(bufA, n, 96, 64, hv);
#pragma unroll
    for (int m = 0; m < 4; ++m) {
      v8f acc = {};
      acc = wmma16(loadAfrag(sW1, m * 16 + n, 96, 0, hv), bf0, acc);
      acc = wmma16(loadAfrag(sW1, m * 16 + n, 96, 32, hv), bf1, acc);
      acc = wmma16(loadAfrag(sW1, m * 16 + n, 96, 64, hv), bf2, acc);
      v8h h;
#pragma unroll
      for (int v = 0; v < 8; ++v) {
        float e = acc[v] + b1[m * 16 + hv * 8 + v];
        h[v] = (_Float16)fmaxf(e, 0.f);
      }
      *(v8h*)(bufB + n * 64 + m * 16 + hv * 8) = h;
    }
  }
  WAVE_LDS_FENCE();

  // ---- layer2: emb2 = relu(W2 @ emb + b2)  (M=64, K=64, N=16) ----
  {
    v16h bf0 = loadBfrag(bufB, n, 64, 0, hv);
    v16h bf1 = loadBfrag(bufB, n, 64, 32, hv);
#pragma unroll
    for (int m = 0; m < 4; ++m) {
      v8f acc = {};
      acc = wmma16(loadAfrag(sW2, m * 16 + n, 64, 0, hv), bf0, acc);
      acc = wmma16(loadAfrag(sW2, m * 16 + n, 64, 32, hv), bf1, acc);
      v8h h;
#pragma unroll
      for (int v = 0; v < 8; ++v) {
        float e = acc[v] + b2[m * 16 + hv * 8 + v];
        h[v] = (_Float16)fmaxf(e, 0.f);
      }
      *(v8h*)(bufA + n * 64 + m * 16 + hv * 8) = h;  // bufA reused (stride 64)
    }
  }
  WAVE_LDS_FENCE();

  // ---- head: rows 0-1 offset, 2-5 routing  (M=16, K=64, N=16) ----
  v8f dor = {};
  {
    v16h bf0 = loadBfrag(bufA, n, 64, 0, hv);
    v16h bf1 = loadBfrag(bufA, n, 64, 32, hv);
    dor = wmma16(loadAfrag(sWOR, n, 64, 0, hv), bf0, dor);
    dor = wmma16(loadAfrag(sWOR, n, 64, 32, hv), bf1, dor);
  }
  // Rows 0..5 live in lanes 0-15 (M = reg index). Broadcast per pixel.
  const float offx = __shfl(dor[0], n, 32) + bo[0];
  const float offy = __shfl(dor[1], n, 32) + bo[1];
  const float r0 = sigmoidf_(__shfl(dor[2], n, 32) + br[0]);
  const float r1 = sigmoidf_(__shfl(dor[3], n, 32) + br[1]);
  const float r2 = sigmoidf_(__shfl(dor[4], n, 32) + br[2]);
  const float r3 = sigmoidf_(__shfl(dor[5], n, 32) + br[3]);

  // ---- fea0: offset bilinear sample, f32 into LDS (aliases bufA/bufB) ----
  WAVE_LDS_FENCE();
  {
    const float gx2 = gx + offx * (2.0f / 127.f);
    const float gy2 = gy + offy * (2.0f / 127.f);
    const float px2 = ((gx2 + 1.0f) * 128.f - 1.0f) * 0.5f;
    const float py2 = ((gy2 + 1.0f) * 128.f - 1.0f) * 0.5f;
    Bilerp t = mk_bilerp(px2, py2, 128, 128);
    const float* xc = xb + (hv * 32) * 16384;
#pragma unroll 4
    for (int i = 0; i < 32; ++i) {
      float v = t.w00 * xc[t.o00] + t.w01 * xc[t.o01] +
                t.w10 * xc[t.o10] + t.w11 * xc[t.o11];
      feaF[n * 64 + hv * 32 + i] = v;
      xc += 16384;
    }
  }
  WAVE_LDS_FENCE();

  // ---- compress: T = [Wc(e0..3) stacked along M] @ fea0, 32x64x16 ----
  v16h cf0 = {}, cf1 = {};
  {
    const float* pf = feaF + n * 64 + hv * 16;
#pragma unroll
    for (int j = 0; j < 16; ++j) cf0[j] = (_Float16)pf[j];
    pf = feaF + n * 64 + 32 + hv * 16;
#pragma unroll
    for (int j = 0; j < 16; ++j) cf1[j] = (_Float16)pf[j];
  }
  v8f t0 = {}, t1 = {};
  t0 = wmma16(loadAfrag(sWC, n, 64, 0, hv), cf0, t0);
  t0 = wmma16(loadAfrag(sWC, n, 64, 32, hv), cf1, t0);
  t1 = wmma16(loadAfrag(sWC, 16 + n, 64, 0, hv), cf0, t1);
  t1 = wmma16(loadAfrag(sWC, 16 + n, 64, 32, hv), cf1, t1);

  // comp[k] = sum_e r_e * T[e*8+k]; rows e0/e2 in lo half, e1/e3 in hi half.
  const float rA = hv ? r1 : r0;
  const float rB = hv ? r3 : r2;
  float comp[8];
#pragma unroll
  for (int v = 0; v < 8; ++v) {
    float s = rA * t0[v] + rB * t1[v];
    comp[v] = s + __shfl_xor(s, 16, 32);
  }

  // ---- expand: out = [We stacked along K] @ (r_e * comp), 64x32x16 ----
  v16h bfe = {};
  const float rE0 = hv ? r2 : r0;  // B element j -> K=j+16*hv: experts 0/1 lo, 2/3 hi
  const float rE1 = hv ? r3 : r1;
#pragma unroll
  for (int j = 0; j < 8; ++j) {
    bfe[j] = (_Float16)(rE0 * comp[j]);
    bfe[j + 8] = (_Float16)(rE1 * comp[j]);
  }
#pragma unroll
  for (int m = 0; m < 4; ++m) {
    v8f acc = {};
    acc = wmma16(loadAfrag(sWE, m * 16 + n, 32, 0, hv), bfe, acc);
#pragma unroll
    for (int v = 0; v < 8; ++v) {
      int ch = m * 16 + hv * 8 + v;
      float val = acc[v] + feaF[n * 64 + ch];  // residual +fea0 in f32
      out[((b * 64 + ch) * 256 + oy) * 256 + ox] = val;
    }
  }
}

extern "C" void kernel_launch(void* const* d_in, const int* in_sizes, int n_in,
                              void* d_out, int out_size, void* d_ws,
                              size_t ws_size, hipStream_t stream) {
  (void)in_sizes; (void)n_in; (void)out_size; (void)d_ws; (void)ws_size;
  scab_fused_gfx1250<<<dim3(4096), dim3(128), 0, stream>>>(
      (const float*)d_in[0],   // x
      (const float*)d_in[1],   // weight_compress
      (const float*)d_in[2],   // weight_expand
      (const float*)d_in[3],   // w1
      (const float*)d_in[4],   // b1
      (const float*)d_in[5],   // w2
      (const float*)d_in[6],   // b2
      (const float*)d_in[7],   // w_route
      (const float*)d_in[8],   // b_route
      (const float*)d_in[9],   // w_off
      (const float*)d_in[10],  // b_off
      (float*)d_out);
}
